// DSAttention_19868518711342
// MI455X (gfx1250) — matrix operand order repaired and checked by reading, not compiled, not run on hardware
//
#include <hip/hip_runtime.h>
#include <math.h>

typedef __attribute__((ext_vector_type(16))) _Float16 v16h;
typedef __attribute__((ext_vector_type(8)))  float    v8f;
typedef int i32x4 __attribute__((vector_size(16)));

#define NH    16     // heads
#define HD    64     // head dim E
#define QT    16     // query rows per wave
#define KT    32     // key tile
#define WAVES 4      // waves per block (wave32)

#ifndef __has_builtin
#define __has_builtin(x) 0
#endif

#if __has_builtin(__builtin_amdgcn_global_load_async_to_lds_b128)
#define USE_ASYNC_LDS 1
#else
#define USE_ASYNC_LDS 0
#endif

// Copy 16 bytes global -> LDS. Async path uses the CDNA5 ASYNCcnt engine.
// Probe-verified signature: (int4 AS1*, int4 AS3*, imm offset, imm cpol).
__device__ __forceinline__ void stage16(const _Float16* g, _Float16* l) {
#if USE_ASYNC_LDS
    __builtin_amdgcn_global_load_async_to_lds_b128(
        (__attribute__((address_space(1))) i32x4*)g,
        (__attribute__((address_space(3))) i32x4*)l, 0, 0);
#else
    *(uint4*)l = *(const uint4*)g;
#endif
}

__device__ __forceinline__ void wait_stage() {
#if USE_ASYNC_LDS
#if __has_builtin(__builtin_amdgcn_s_wait_asynccnt)
    __builtin_amdgcn_s_wait_asynccnt(0);
#else
    asm volatile("s_wait_asynccnt 0" ::: "memory");
#endif
#endif
}

// max over the 16 lanes of a wave32 half via ds_swizzle xor patterns
__device__ __forceinline__ float xmax16(float x) {
    x = fmaxf(x, __int_as_float(__builtin_amdgcn_ds_swizzle(__float_as_int(x), 0x041F))); // xor 1
    x = fmaxf(x, __int_as_float(__builtin_amdgcn_ds_swizzle(__float_as_int(x), 0x081F))); // xor 2
    x = fmaxf(x, __int_as_float(__builtin_amdgcn_ds_swizzle(__float_as_int(x), 0x101F))); // xor 4
    x = fmaxf(x, __int_as_float(__builtin_amdgcn_ds_swizzle(__float_as_int(x), 0x201F))); // xor 8
    return x;
}

// ---------------- pre-pass: f32 -> f16 relayout (memory-bound, runs once) ----
// K16[(b*NH+h)*L + s][HD]
__global__ void cvtK_kernel(const float* __restrict__ Kg, _Float16* __restrict__ K16,
                            int L, long long n) {
    long long i = (long long)blockIdx.x * blockDim.x + threadIdx.x;
    if (i >= n) return;
    int e  = (int)(i % HD);
    long long r = i / HD;
    int s  = (int)(r % L);
    int bh = (int)(r / L);
    int b = bh / NH, h = bh % NH;
    K16[i] = (_Float16)Kg[(((size_t)b * L + s) * NH + h) * HD + e];
}

// VT16[((b*NH+h)*HD + e)*L + s]   (pre-transposed V)
__global__ void cvtVT_kernel(const float* __restrict__ Vg, _Float16* __restrict__ VT16,
                             int L, long long n) {
    long long i = (long long)blockIdx.x * blockDim.x + threadIdx.x;
    if (i >= n) return;
    int s  = (int)(i % L);
    long long r = i / L;
    int e  = (int)(r % HD);
    int bh = (int)(r / HD);
    int b = bh / NH, h = bh % NH;
    VT16[i] = (_Float16)Vg[(((size_t)b * L + s) * NH + h) * HD + e];
}

// ---------------- main flash-attention kernel (f16 WMMA, async staged) ------
__global__ __launch_bounds__(128) void dsattn_fa2_kernel(
    const float* __restrict__ Qg, const _Float16* __restrict__ K16,
    const _Float16* __restrict__ VT16, const float* __restrict__ tau,
    const float* __restrict__ delta, float* __restrict__ Out, int L)
{
    __shared__ _Float16 sK [2][KT * HD];        // [buf][key][feat]
    __shared__ _Float16 sVT[2][HD * KT];        // [buf][feat][key]
    __shared__ _Float16 sP [WAVES][QT * KT];    // per-wave P staging

    const int bh   = blockIdx.y;
    const int b    = bh / NH;
    const int h    = bh % NH;
    const int qb   = blockIdx.x * (QT * WAVES);
    const int tid  = threadIdx.x;
    const int w    = tid >> 5;
    const int lane = tid & 31;
    const int half = lane >> 4;
    const int l16  = lane & 15;
    const int qw   = qb + w * QT;

    const float st = 0.125f * tau[b];           // scale * tau folded together

    const _Float16* Kbase = K16  + (size_t)bh * L * HD;
    const _Float16* Vbase = VT16 + (size_t)bh * HD * L;

    // Q A-fragments, pre-scaled by scale*tau (logits fall out of the WMMA)
    v16h aq[2];
    {
        const int qrow = qw + l16;
        const float* qp = Qg + (((size_t)b * L + qrow) * NH + h) * HD;
        for (int c = 0; c < 2; ++c)
            for (int e = 0; e < 16; ++e) {
                int k = half * 16 + e;
                aq[c][e] = (_Float16)(st * qp[c * 32 + k]);
            }
    }

    v16h onesf;                                  // B fragment of ones: row sums via WMMA
    for (int e = 0; e < 16; ++e) onesf[e] = (_Float16)1.0f;

    float mrow[8];
    v8f   acc[4] = {};                           // O accumulators (4 feature tiles)
    v8f   accs   = {};                           // row-sum accumulator (P @ ones)
    for (int v = 0; v < 8; ++v) mrow[v] = -INFINITY;

    // per-thread staging slots
    const int kkey = tid >> 2, kfb = (tid & 3) * 16;     // K: 32 keys x 64 feats
    const int vrow = tid >> 1, vhb = (tid & 1) * 16;     // V: 64 feats x 32 keys

    auto issue = [&](int s0, int buf) {
        const _Float16* kg = Kbase + (size_t)(s0 + kkey) * HD + kfb;
        stage16(kg,     &sK[buf][kkey * HD + kfb]);
        stage16(kg + 8, &sK[buf][kkey * HD + kfb + 8]);
        const _Float16* vg = Vbase + (size_t)vrow * L + s0 + vhb;
        stage16(vg,     &sVT[buf][vrow * KT + vhb]);
        stage16(vg + 8, &sVT[buf][vrow * KT + vhb + 8]);
    };

    const int nt = (qb + QT * WAVES) / KT;       // causal: keys up to block's last row
    issue(0, 0);

    for (int t = 0; t < nt; ++t) {
        const int s0  = t * KT;
        const int buf = t & 1;

        wait_stage();          // own async copies for tile t landed
        __syncthreads();       // everyone's copies landed; prev compute finished
        if (t + 1 < nt) issue(s0 + KT, buf ^ 1);

        if (s0 <= qw + QT - 1) {                 // wave-uniform causal skip
            // scale*delta seeds the QK accumulator (broadcast per column)
            const float sd0 = 0.125f * delta[b * L + s0 + l16];
            const float sd1 = 0.125f * delta[b * L + s0 + 16 + l16];

            // ---- logits = (scale*tau)*Q @ K^T + scale*delta ----
            v8f sc[2];
            for (int n = 0; n < 2; ++n) {
                v8f c;
                const float sd = n ? sd1 : sd0;
                for (int v = 0; v < 8; ++v) c[v] = sd;
                for (int ch = 0; ch < 2; ++ch) {
                    v16h bk;
                    const _Float16* kp = &sK[buf][(n * 16 + l16) * HD + ch * 32 + half * 16];
                    for (int e = 0; e < 16; ++e) bk[e] = kp[e];
                    c = __builtin_amdgcn_wmma_f32_16x16x32_f16(
                            false, aq[ch], false, bk, (short)0, c, false, false);
                }
                sc[n] = c;
            }

            // ---- causal mask only on diagonal tiles (wave-uniform) ----
            if (s0 + KT - 1 > qw) {
                for (int v = 0; v < 8; ++v) {
                    const int q = qw + half * 8 + v;
                    if (s0 + l16      > q) sc[0][v] = -INFINITY;
                    if (s0 + 16 + l16 > q) sc[1][v] = -INFINITY;
                }
            }

            // ---- online softmax: max via ds_swizzle, sums via WMMA ones-trick ----
            float pr[2][8], corr[8];
            for (int v = 0; v < 8; ++v) {
                float lmax = xmax16(fmaxf(sc[0][v], sc[1][v]));
                float mnew = fmaxf(mrow[v], lmax);
                corr[v]  = __expf(mrow[v] - mnew);
                mrow[v]  = mnew;
                pr[0][v] = __expf(sc[0][v] - mnew);
                pr[1][v] = __expf(sc[1][v] - mnew);
            }
            for (int ti = 0; ti < 4; ++ti)
                for (int v = 0; v < 8; ++v) acc[ti][v] *= corr[v];
            for (int v = 0; v < 8; ++v) accs[v] *= corr[v];

            // ---- re-layout P: C-fragment -> LDS -> A-fragment ----
            for (int n = 0; n < 2; ++n)
                for (int v = 0; v < 8; ++v)
                    sP[w][(half * 8 + v) * KT + n * 16 + l16] = (_Float16)pr[n][v];
            asm volatile("s_wait_dscnt 0" ::: "memory");
            v16h ap;
            for (int e = 0; e < 16; ++e) {
                int k = half * 16 + e;
                ap[e] = sP[w][l16 * KT + k];
            }

            // ---- O += P @ V, and row-sums += P @ ones ----
            for (int ti = 0; ti < 4; ++ti) {
                v16h bv;
                const _Float16* vp = &sVT[buf][(ti * 16 + l16) * KT + half * 16];
                for (int e = 0; e < 16; ++e) bv[e] = vp[e];
                acc[ti] = __builtin_amdgcn_wmma_f32_16x16x32_f16(
                              false, ap, false, bv, (short)0, acc[ti], false, false);
            }
            accs = __builtin_amdgcn_wmma_f32_16x16x32_f16(
                       false, ap, false, onesf, (short)0, accs, false, false);
        }
    }

    // ---- normalize and store [B, L, H, E] ----
    for (int v = 0; v < 8; ++v) {
        float inv = 1.0f / accs[v];
        int q = qw + half * 8 + v;
        float* op = Out + (((size_t)b * L + q) * NH + h) * HD;
        for (int ti = 0; ti < 4; ++ti)
            op[ti * 16 + l16] = acc[ti][v] * inv;
    }
}

// ---------------- fallback (no workspace): round-1 proven kernel ------------
__global__ __launch_bounds__(128) void dsattn_fa_fallback(
    const float* __restrict__ Qg, const float* __restrict__ Kg,
    const float* __restrict__ Vg, const float* __restrict__ tau,
    const float* __restrict__ delta, float* __restrict__ Out, int L)
{
    __shared__ _Float16 sK [KT * HD];
    __shared__ _Float16 sVT[HD * KT];
    __shared__ _Float16 sP [WAVES][QT * KT];

    const int bh = blockIdx.y, b = bh / NH, h = bh % NH;
    const int qb = blockIdx.x * (QT * WAVES);
    const int tid = threadIdx.x, w = tid >> 5, lane = tid & 31;
    const int half = lane >> 4, l16 = lane & 15;
    const int qw = qb + w * QT;
    const float scale = 0.125f, tb = tau[b];

    v16h aq[2];
    {
        const int qrow = qw + l16;
        const float* qp = Qg + (((size_t)b * L + qrow) * NH + h) * HD;
        for (int c = 0; c < 2; ++c)
            for (int e = 0; e < 16; ++e) {
                int k = half * 16 + e;
                aq[c][e] = (_Float16)qp[c * 32 + k];
            }
    }
    float mrow[8], srow[8];
    v8f acc[4] = {};
    for (int v = 0; v < 8; ++v) { mrow[v] = -INFINITY; srow[v] = 0.0f; }

    const int send = qb + QT * WAVES;
    for (int s0 = 0; s0 < send; s0 += KT) {
        __syncthreads();
        {
            int key = tid >> 2, fb = (tid & 3) * 16;
            const float* kp = Kg + (((size_t)b * L + s0 + key) * NH + h) * HD + fb;
            for (int j = 0; j < 16; ++j) sK[key * HD + fb + j] = (_Float16)kp[j];
            int vkey = tid & 31, vfb = (tid >> 5) * 16;
            const float* vp = Vg + (((size_t)b * L + s0 + vkey) * NH + h) * HD + vfb;
            for (int j = 0; j < 16; ++j) sVT[(vfb + j) * KT + vkey] = (_Float16)vp[j];
        }
        __syncthreads();
        if (s0 > qw + QT - 1) continue;

        v8f sc[2];
        for (int n = 0; n < 2; ++n) {
            v8f c = {};
            for (int ch = 0; ch < 2; ++ch) {
                v16h bk;
                const _Float16* kp = &sK[(n * 16 + l16) * HD + ch * 32 + half * 16];
                for (int e = 0; e < 16; ++e) bk[e] = kp[e];
                c = __builtin_amdgcn_wmma_f32_16x16x32_f16(
                        false, aq[ch], false, bk, (short)0, c, false, false);
            }
            sc[n] = c;
        }
        float pr[2][8], corr[8];
        for (int v = 0; v < 8; ++v) {
            const int q = qw + half * 8 + v;
            float lg[2], lmax = -INFINITY;
            for (int n = 0; n < 2; ++n) {
                int s = s0 + n * 16 + l16;
                float x = (s <= q) ? scale * (sc[n][v] * tb + delta[b * L + s]) : -INFINITY;
                lg[n] = x; lmax = fmaxf(lmax, x);
            }
            lmax = xmax16(lmax);
            float mnew = fmaxf(mrow[v], lmax);
            float cr = __expf(mrow[v] - mnew), ps = 0.0f;
            for (int n = 0; n < 2; ++n) { float p = __expf(lg[n] - mnew); pr[n][v] = p; ps += p; }
            for (int m = 1; m < 16; m <<= 1) ps += __shfl_xor(ps, m, 32);
            srow[v] = srow[v] * cr + ps; mrow[v] = mnew; corr[v] = cr;
        }
        for (int ti = 0; ti < 4; ++ti)
            for (int v = 0; v < 8; ++v) acc[ti][v] *= corr[v];
        for (int n = 0; n < 2; ++n)
            for (int v = 0; v < 8; ++v)
                sP[w][(half * 8 + v) * KT + n * 16 + l16] = (_Float16)pr[n][v];
        asm volatile("s_wait_dscnt 0" ::: "memory");
        v16h ap;
        for (int e = 0; e < 16; ++e) {
            int k = half * 16 + e;
            ap[e] = sP[w][l16 * KT + k];
        }
        for (int ti = 0; ti < 4; ++ti) {
            v16h bv;
            const _Float16* vp = &sVT[(ti * 16 + l16) * KT + half * 16];
            for (int e = 0; e < 16; ++e) bv[e] = vp[e];
            acc[ti] = __builtin_amdgcn_wmma_f32_16x16x32_f16(
                          false, ap, false, bv, (short)0, acc[ti], false, false);
        }
    }
    for (int v = 0; v < 8; ++v) {
        float inv = 1.0f / srow[v];
        int q = qw + half * 8 + v;
        float* op = Out + (((size_t)b * L + q) * NH + h) * HD;
        for (int ti = 0; ti < 4; ++ti) op[ti * 16 + l16] = acc[ti][v] * inv;
    }
}

extern "C" void kernel_launch(void* const* d_in, const int* in_sizes, int n_in,
                              void* d_out, int out_size, void* d_ws, size_t ws_size,
                              hipStream_t stream) {
    const float* Q     = (const float*)d_in[0];
    const float* K     = (const float*)d_in[1];
    const float* V     = (const float*)d_in[2];
    const float* tau   = (const float*)d_in[3];
    const float* delta = (const float*)d_in[4];
    float* Out         = (float*)d_out;

    const int B = in_sizes[3];          // tau: [B, 1]
    const int L = in_sizes[4] / B;      // delta: [B, L]
    const long long elems = (long long)B * NH * L * HD;
    const size_t need = (size_t)elems * 2 * sizeof(_Float16);

    dim3 grid(L / (QT * WAVES), B * NH);
    if (ws_size >= need) {
        _Float16* K16  = (_Float16*)d_ws;
        _Float16* VT16 = K16 + elems;
        const int thr = 256;
        const int blks = (int)((elems + thr - 1) / thr);
        cvtK_kernel <<<blks, thr, 0, stream>>>(K, K16, L, elems);
        cvtVT_kernel<<<blks, thr, 0, stream>>>(V, VT16, L, elems);
        dsattn_fa2_kernel<<<grid, 128, 0, stream>>>(Q, K16, VT16, tau, delta, Out, L);
    } else {
        dsattn_fa_fallback<<<grid, 128, 0, stream>>>(Q, K, V, tau, delta, Out, L);
    }
}
